// CertBP_91250875171590
// MI455X (gfx1250) — matrix operand order repaired and checked by reading, not compiled, not run on hardware
//
#include <hip/hip_runtime.h>
#include <hip/hip_fp16.h>
#include <math.h>

typedef __attribute__((ext_vector_type(16))) _Float16     v16h;
typedef __attribute__((ext_vector_type(2)))  _Float16     h2;
typedef __attribute__((ext_vector_type(8)))  float        v8f;
typedef __attribute__((ext_vector_type(8)))  unsigned int v8u;
typedef __attribute__((ext_vector_type(4)))  unsigned int u32x4;
typedef int v4i_ __attribute__((vector_size(16)));

#define EPS_BP   1e-12f
#define W_MAX_BP 0.8f
#define ALPHA_MX 1.5f
#define T_BP     10
#define ETA_BP   0.2f

#if __has_builtin(__builtin_amdgcn_global_load_async_to_lds_b128) && __has_builtin(__builtin_amdgcn_s_wait_asynccnt)
#define USE_ASYNC_LDS 1
#define GPTR128(p) ((__attribute__((address_space(1))) v4i_*)(uintptr_t)(p))
#define LPTR128(p) ((__attribute__((address_space(3))) v4i_*)(p))
#else
#define USE_ASYNC_LDS 0
#endif

static __device__ __forceinline__ float sigmoidf_(float x){ return 1.0f/(1.0f+expf(-x)); }
static __device__ __forceinline__ float softplusf_(float x){ return (x>20.f)?x:log1pf(expf(x)); }
// ISA 16-bit A-fragment K mapping within a 32-wide K block (per VGPR r, low lane-half)
static __device__ __forceinline__ int kmap(int r){ return (r < 4) ? (r << 1) : (16 + ((r - 4) << 1)); }

// ---------------- utility ----------------
__global__ void k_zero(float* __restrict__ p, int n){
  int i = blockIdx.x*blockDim.x + threadIdx.x;
  if (i < n) p[i] = 0.f;
}

// ---------------- pack x (f32) into WMMA A-fragment order (f16) ----------------
// Ap[((mT*16 + kkIdx)*32 + lane)*16 + el]
__global__ void k_pack_a(const float* __restrict__ x, _Float16* __restrict__ Ap, int N, int total){
  int idx = blockIdx.x*blockDim.x + threadIdx.x;
  if (idx >= total) return;
  int el = idx & 15, lane = (idx >> 4) & 31, kkIdx = (idx >> 9) & 15, mT = idx >> 13;
  int m = mT*16 + (lane & 15); if (m >= N) m = N - 1;
  int r = el >> 1, p = el & 1;
  int k = kkIdx*32 + kmap(r) + (lane >> 4)*8 + p;
  Ap[idx] = (_Float16)x[(size_t)m*512 + k];
}

// ---------------- pack a [K,NCOL] f32 weight into WMMA B-fragment order (f16) ----------------
// Bp[((nT*(K/32) + kkIdx)*32 + lane)*16 + el]; element el <-> K row kk + (el>>3)*16 + half*8 + (el&7)
__global__ void k_pack_b(const float* __restrict__ W, _Float16* __restrict__ Bp,
                         int K, int NCOL, int total){
  int idx = blockIdx.x*blockDim.x + threadIdx.x;
  if (idx >= total) return;
  int el = idx & 15, lane = (idx >> 4) & 31;
  int kkN = K >> 5;
  int kkIdx = (idx >> 9) % kkN;
  int nT = idx / (512 * kkN);
  int col = nT*16 + (lane & 15);
  int k = kkIdx*32 + (el >> 3)*16 + (lane >> 4)*8 + (el & 7);
  Bp[idx] = (_Float16)W[(size_t)k*NCOL + col];
}

// ---------------- transpose enc_w2 [256,8] -> w2t [8,256] ----------------
__global__ void k_pack_w2t(const float* __restrict__ w2, float* __restrict__ w2t, int n){
  int i = blockIdx.x*blockDim.x + threadIdx.x;
  if (i >= n) return;
  int c = i >> 8, k = i & 255;
  w2t[i] = w2[k*8 + c];
}

// ---------------- encoder: h = relu(x @ W1 + b1), f16 out ----------------
__global__ void __launch_bounds__(32)
k_encoder(const _Float16* __restrict__ Ap, const _Float16* __restrict__ Bp,
          const float* __restrict__ b1, _Float16* __restrict__ hh, int N){
  const int lane = threadIdx.x, half = lane >> 4, lm = lane & 15;
  const int mT = blockIdx.x, nT = blockIdx.y;
  const v8u* A32 = (const v8u*)Ap;
  const v8u* B32 = (const v8u*)Bp;
  v8f acc = {};
  #pragma unroll 4
  for (int kkIdx = 0; kkIdx < 16; kkIdx++){
    v16h a = __builtin_bit_cast(v16h, A32[(size_t)(mT*16 + kkIdx)*32 + lane]);
    v16h b = __builtin_bit_cast(v16h, B32[(size_t)(nT*16 + kkIdx)*32 + lane]);
    acc = __builtin_amdgcn_wmma_f32_16x16x32_f16(false, a, false, b, (short)0, acc, false, false);
  }
  const int col = nT*16 + lm;
  const float bn = b1[col];
  const int mBase = mT*16;
  if (mBase + 16 <= N){
    #pragma unroll
    for (int r = 0; r < 8; r++){
      float v = acc[r] + bn;
      hh[(size_t)(mBase + r + half*8)*256 + col] = (_Float16)(v > 0.f ? v : 0.f);
    }
  } else {
    #pragma unroll
    for (int r = 0; r < 8; r++){
      int m = mBase + r + half*8;
      if (m < N){
        float v = acc[r] + bn;
        hh[(size_t)m*256 + col] = (_Float16)(v > 0.f ? v : 0.f);
      }
    }
  }
}

// ---------------- logits -> log_softmax + phi ----------------
__global__ void k_logits(const _Float16* __restrict__ hh, const float* __restrict__ w2t,
                         const float* __restrict__ b2, float* __restrict__ log_phi,
                         float* __restrict__ phi, int N){
  int t = blockIdx.x*blockDim.x + threadIdx.x;
  int node = t >> 3, c = t & 7;
  if (node >= N) return;
  const u32x4* hq = (const u32x4*)(hh + (size_t)node*256);
  const float* wr = w2t + c*256;
  float acc = b2[c];
  for (int q = 0; q < 32; q++){
    u32x4 hv = hq[q];
    #pragma unroll
    for (int j = 0; j < 4; j++){
      h2 hp = __builtin_bit_cast(h2, hv[j]);
      int k = q*8 + j*2;
      acc += (float)hp.x * wr[k] + (float)hp.y * wr[k+1];
    }
  }
  float mx = acc;
  #pragma unroll
  for (int off = 4; off; off >>= 1) mx = fmaxf(mx, __shfl_xor(mx, off, 8));
  float e = expf(acc - mx), s = e;
  #pragma unroll
  for (int off = 4; off; off >>= 1) s += __shfl_xor(s, off, 8);
  float lp = acc - mx - logf(s);
  log_phi[t] = lp;
  phi[t] = expf(lp);
}

// ---------------- degree ----------------
__global__ void k_degree(const int* __restrict__ src, float* __restrict__ deg, int E){
  int i = blockIdx.x*blockDim.x + threadIdx.x;
  if (i < E) atomicAdd(&deg[src[i]], 1.0f);
}
__global__ void k_logdeg(const float* __restrict__ deg, float* __restrict__ ld, int N){
  int i = blockIdx.x*blockDim.x + threadIdx.x;
  if (i < N) ld[i] = logf(deg[i] + 1.0f);
}

// ---------------- edge MLP ----------------
// one wave handles 16 edges; h rows staged via async global->LDS; A-frag built with packed f16 math
__global__ void __launch_bounds__(32)
k_edge_mlp(const _Float16* __restrict__ hh, const int* __restrict__ src, const int* __restrict__ dst,
           const float* __restrict__ ld, const _Float16* __restrict__ BpE,
           const float* __restrict__ ew1, const float* __restrict__ b1e,
           const float* __restrict__ w2e, const float* __restrict__ b2e,
           float* __restrict__ wp, int E){
  __shared__ _Float16 hsL[16*256];
  __shared__ _Float16 hdL[16*256];
  __shared__ float s0L[16], s1L[16];
  __shared__ int sNode[16], dNode[16];
  const int lane = threadIdx.x;
  const int half = lane >> 4, lm = lane & 15;
  const int eBase = blockIdx.x * 16;
  if (lane < 16){
    int e = eBase + lane; if (e >= E) e = E - 1;
    int s = src[e], d = dst[e];
    sNode[lane] = s; dNode[lane] = d;
    float la = ld[s], lb = ld[d];
    s0L[lane] = la + lb;
    s1L[lane] = fabsf(la - lb);
  }
  __syncthreads();
  // stage 16 src + 16 dst rows of h (512 B each) into LDS
#if USE_ASYNC_LDS
  for (int t = lane; t < 1024; t += 32){
    int mat = t >> 9, e = (t >> 5) & 15, ch = t & 31;
    int node = mat ? dNode[e] : sNode[e];
    const char* g = (const char*)hh + (size_t)node*512 + (size_t)ch*16;
    _Float16* l = (mat ? hdL : hsL) + e*256 + ch*8;
    __builtin_amdgcn_global_load_async_to_lds_b128(GPTR128(g), LPTR128(l), 0, 0);
  }
  __builtin_amdgcn_s_wait_asynccnt(0);
#else
  for (int t = lane; t < 1024; t += 32){
    int mat = t >> 9, e = (t >> 5) & 15, ch = t & 31;
    int node = mat ? dNode[e] : sNode[e];
    u32x4 v = *(const u32x4*)((const char*)hh + (size_t)node*512 + (size_t)ch*16);
    u32x4* l = (u32x4*)((mat ? hdL : hsL) + e*256 + ch*8);
    *l = v;
  }
#endif
  __syncthreads();

  const v8u* B32 = (const v8u*)BpE;
  v8f acc[4];
  #pragma unroll
  for (int t = 0; t < 4; t++){
    #pragma unroll
    for (int r = 0; r < 8; r++) acc[t][r] = 0.f;
  }

  for (int kkIdx = 0; kkIdx < 16; kkIdx++){
    const int kk = kkIdx*32;
    v8u au;
    const int koff = half*8;
    #pragma unroll
    for (int r = 0; r < 8; r++){
      int kb = kk + kmap(r) + koff;
      if (kb < 256){
        h2 a_ = *(const h2*)&hsL[lm*256 + kb];
        h2 b_ = *(const h2*)&hdL[lm*256 + kb];
        h2 p  = a_ * b_;                       // v_pk_mul_f16
        au[r] = __builtin_bit_cast(unsigned int, p);
      } else {
        int k2 = kb - 256;
        h2 a_ = *(const h2*)&hsL[lm*256 + k2];
        h2 b_ = *(const h2*)&hdL[lm*256 + k2];
        h2 d  = a_ - b_;                       // v_pk_add_f16
        au[r] = __builtin_bit_cast(unsigned int, d) & 0x7FFF7FFFu;  // packed |.|
      }
    }
    v16h a = __builtin_bit_cast(v16h, au);
    #pragma unroll
    for (int t = 0; t < 4; t++){
      v16h b = __builtin_bit_cast(v16h, B32[(size_t)(t*16 + kkIdx)*32 + lane]);
      acc[t] = __builtin_amdgcn_wmma_f32_16x16x32_f16(false, a, false, b, (short)0, acc[t], false, false);
    }
  }

  // fuse struct features (rows 512,513 of edge_w1, kept f32), bias, relu, dot with edge_w2
  float partial[8];
  #pragma unroll
  for (int r = 0; r < 8; r++) partial[r] = 0.f;
  #pragma unroll
  for (int t = 0; t < 4; t++){
    int n = t*16 + lm;
    float w512 = ew1[(size_t)512*64 + n];
    float w513 = ew1[(size_t)513*64 + n];
    float bn = b1e[n], w2n = w2e[n];
    #pragma unroll
    for (int r = 0; r < 8; r++){
      int mrow = r + half*8;
      float v = acc[t][r] + s0L[mrow]*w512 + s1L[mrow]*w513 + bn;
      v = v > 0.f ? v : 0.f;
      partial[r] += v * w2n;
    }
  }
  #pragma unroll
  for (int r = 0; r < 8; r++){
    #pragma unroll
    for (int off = 1; off < 16; off <<= 1)
      partial[r] += __shfl_xor(partial[r], off, 16);
  }
  if (lm == 0){
    float bb = b2e[0];
    #pragma unroll
    for (int r = 0; r < 8; r++){
      int e = eBase + r + half*8;
      if (e < E) wp[e] = W_MAX_BP * sigmoidf_(partial[r] + bb);
    }
  }
}

// ---------------- symmetrize w, edge_norm ----------------
__global__ void k_edge_sym(const float* __restrict__ wp, const int* __restrict__ rev,
                           const int* __restrict__ src, const int* __restrict__ dst,
                           const float* __restrict__ deg, float* __restrict__ w,
                           float* __restrict__ en, int E){
  int e = blockIdx.x*blockDim.x + threadIdx.x;
  if (e >= E) return;
  w[e] = 0.5f * (wp[e] + wp[rev[e]]);
  float ds = fmaxf(deg[src[e]], 1.f), dd = fmaxf(deg[dst[e]], 1.f);
  en[e] = rsqrtf(ds * dd);
}

// ---------------- R symmetrize + scale*tanh ----------------
__global__ void k_Rsym(const float* __restrict__ R_raw, const float* __restrict__ R_slog,
                       float* __restrict__ Rs){
  int i = threadIdx.x;          // 64 threads
  int c = i >> 3, d = i & 7;
  float r = 0.5f * (R_raw[c*8 + d] + R_raw[d*8 + c]);
  float sc = softplusf_(R_slog[0]) + 1e-6f;
  Rs[i] = sc * tanhf(r);
}

// ---------------- m init ----------------
__global__ void k_m_init(const float* __restrict__ phi, const int* __restrict__ src,
                         float* __restrict__ m, int E){
  int t = blockIdx.x*blockDim.x + threadIdx.x;
  if (t >= E*8) return;
  int e = t >> 3, c = t & 7;
  m[t] = phi[src[e]*8 + c];
}

// ---------------- per-edge factor message + scatter ----------------
__global__ void k_edge_f(const float* __restrict__ m, const float* __restrict__ w,
                         const float* __restrict__ en, const int* __restrict__ dst,
                         const float* __restrict__ Rs, float* __restrict__ lf,
                         float* __restrict__ sum_in, int E){
  __shared__ float Rsh[64];
  if (threadIdx.x < 64) Rsh[threadIdx.x] = Rs[threadIdx.x];
  __syncthreads();
  int e = blockIdx.x*blockDim.x + threadIdx.x;
  if (e >= E) return;
  float mm[8], f[8];
  #pragma unroll
  for (int c = 0; c < 8; c++) mm[c] = m[(size_t)e*8 + c];
  float we = w[e], ene = en[e];
  #pragma unroll
  for (int d = 0; d < 8; d++) f[d] = 0.f;
  #pragma unroll
  for (int c = 0; c < 8; c++){
    float mc = mm[c];
    #pragma unroll
    for (int d = 0; d < 8; d++) f[d] += mc * expf(we * Rsh[c*8 + d]);
  }
  int dn = dst[e];
  #pragma unroll
  for (int d = 0; d < 8; d++){
    float l = logf(fmaxf(f[d], EPS_BP)) * ene;
    lf[(size_t)e*8 + d] = l;
    atomicAdd(&sum_in[dn*8 + d], l);
  }
}

// ---------------- message update ----------------
__global__ void k_edge_m(float* __restrict__ m, const float* __restrict__ lf,
                         const float* __restrict__ sum_in, const float* __restrict__ log_phi,
                         const int* __restrict__ src, const int* __restrict__ rev,
                         const float* __restrict__ msg_logit, int E){
  int e = blockIdx.x*blockDim.x + threadIdx.x;
  if (e >= E) return;
  int s = src[e], re = rev[e];
  __builtin_prefetch(&lf[(size_t)re*8], 0, 1);   // global_prefetch
  float alpha = ALPHA_MX * sigmoidf_(msg_logit[0]);
  float t[8]; float mx = -1e30f;
  #pragma unroll
  for (int d = 0; d < 8; d++){
    float excl = sum_in[s*8 + d] - lf[(size_t)re*8 + d];
    t[d] = log_phi[s*8 + d] + alpha * excl;
    mx = fmaxf(mx, t[d]);
  }
  float ssum = 0.f;
  #pragma unroll
  for (int d = 0; d < 8; d++){ t[d] = expf(t[d] - mx); ssum += t[d]; }
  float inv = 1.f / ssum;
  float nm[8]; float tot = 0.f;
  #pragma unroll
  for (int d = 0; d < 8; d++){
    float mv = (1.f - ETA_BP) * m[(size_t)e*8 + d] + ETA_BP * t[d] * inv;
    mv = fmaxf(mv, EPS_BP);
    nm[d] = mv; tot += mv;
  }
  float it = 1.f / tot;
  #pragma unroll
  for (int d = 0; d < 8; d++) m[(size_t)e*8 + d] = nm[d] * it;
}

// ---------------- final beliefs ----------------
__global__ void k_beliefs(const float* __restrict__ log_phi, const float* __restrict__ sum_in,
                          const float* __restrict__ msg_logit, float* __restrict__ out, int N){
  int n = blockIdx.x*blockDim.x + threadIdx.x;
  if (n >= N) return;
  float alpha = ALPHA_MX * sigmoidf_(msg_logit[0]);
  float t[8]; float mx = -1e30f;
  #pragma unroll
  for (int d = 0; d < 8; d++){
    t[d] = log_phi[n*8 + d] + alpha * sum_in[n*8 + d];
    mx = fmaxf(mx, t[d]);
  }
  float s = 0.f;
  #pragma unroll
  for (int d = 0; d < 8; d++){ t[d] = expf(t[d] - mx); s += t[d]; }
  float inv = 1.f / s;
  #pragma unroll
  for (int d = 0; d < 8; d++) out[n*8 + d] = t[d] * inv;
}

// ---------------- launch ----------------
extern "C" void kernel_launch(void* const* d_in, const int* in_sizes, int n_in,
                              void* d_out, int out_size, void* d_ws, size_t ws_size,
                              hipStream_t stream){
  (void)n_in; (void)out_size; (void)ws_size;
  const float* x       = (const float*)d_in[0];
  const int*   ei      = (const int*)d_in[1];
  const int*   rev     = (const int*)d_in[2];
  const float* enc_w1  = (const float*)d_in[3];
  const float* enc_b1  = (const float*)d_in[4];
  const float* enc_w2  = (const float*)d_in[5];
  const float* enc_b2  = (const float*)d_in[6];
  const float* edge_w1 = (const float*)d_in[7];
  const float* edge_b1 = (const float*)d_in[8];
  const float* edge_w2 = (const float*)d_in[9];
  const float* edge_b2 = (const float*)d_in[10];
  const float* R_raw   = (const float*)d_in[11];
  const float* R_slog  = (const float*)d_in[12];
  const float* msg_lg  = (const float*)d_in[13];

  const int N = in_sizes[0] / 512;
  const int E = in_sizes[1] / 2;
  const int* src = ei;
  const int* dst = ei + E;
  auto cdiv = [](int a, int b){ return (a + b - 1) / b; };
  const int mTiles = cdiv(N, 16);
  const int Npad = mTiles * 16;

  char* wsp = (char*)d_ws;
  auto alloc = [&](size_t b)->void*{ void* p = (void*)wsp; wsp += (b + 255) & ~(size_t)255; return p; };
  _Float16* Apack  = (_Float16*)alloc((size_t)Npad*512*2);
  _Float16* BpEnc  = (_Float16*)alloc((size_t)512*256*2);
  _Float16* BpEdge = (_Float16*)alloc((size_t)512*64*2);
  _Float16* hh     = (_Float16*)alloc((size_t)N*256*2);
  float* w2t     = (float*)alloc((size_t)2048*4);
  float* log_phi = (float*)alloc((size_t)N*8*4);
  float* phi     = (float*)alloc((size_t)N*8*4);
  float* deg     = (float*)alloc((size_t)N*4);
  float* ldg     = (float*)alloc((size_t)N*4);
  float* wpre    = (float*)alloc((size_t)E*4);
  float* w       = (float*)alloc((size_t)E*4);
  float* en      = (float*)alloc((size_t)E*4);
  float* Rs      = (float*)alloc(64*4);
  float* m       = (float*)alloc((size_t)E*8*4);
  float* lf      = (float*)alloc((size_t)E*8*4);
  float* sum_in  = (float*)alloc((size_t)N*8*4);

  const int aTot  = Npad*512;
  const int bTotE = 16*16*512;   // enc_w1: (256/16)*(512/32)*512
  const int bTotD = 4*16*512;    // edge_w1[:512]: (64/16)*(512/32)*512

  k_pack_a<<<cdiv(aTot,256),256,0,stream>>>(x, Apack, N, aTot);
  k_pack_b<<<cdiv(bTotE,256),256,0,stream>>>(enc_w1, BpEnc, 512, 256, bTotE);
  k_pack_b<<<cdiv(bTotD,256),256,0,stream>>>(edge_w1, BpEdge, 512, 64, bTotD);
  k_pack_w2t<<<cdiv(2048,256),256,0,stream>>>(enc_w2, w2t, 2048);
  k_zero<<<cdiv(N,256),256,0,stream>>>(deg, N);

  dim3 gEnc(mTiles, 16);
  k_encoder<<<gEnc, 32, 0, stream>>>(Apack, BpEnc, enc_b1, hh, N);
  k_logits<<<cdiv(N*8,256),256,0,stream>>>(hh, w2t, enc_b2, log_phi, phi, N);
  k_degree<<<cdiv(E,256),256,0,stream>>>(src, deg, E);
  k_logdeg<<<cdiv(N,256),256,0,stream>>>(deg, ldg, N);
  k_edge_mlp<<<cdiv(E,16), 32, 0, stream>>>(hh, src, dst, ldg, BpEdge, edge_w1,
                                            edge_b1, edge_w2, edge_b2, wpre, E);
  k_edge_sym<<<cdiv(E,256),256,0,stream>>>(wpre, rev, src, dst, deg, w, en, E);
  k_Rsym<<<1,64,0,stream>>>(R_raw, R_slog, Rs);
  k_m_init<<<cdiv(E*8,256),256,0,stream>>>(phi, src, m, E);

  for (int it = 0; it < T_BP; ++it){
    k_zero<<<cdiv(N*8,256),256,0,stream>>>(sum_in, N*8);
    k_edge_f<<<cdiv(E,256),256,0,stream>>>(m, w, en, dst, Rs, lf, sum_in, E);
    k_edge_m<<<cdiv(E,256),256,0,stream>>>(m, lf, sum_in, log_phi, src, rev, msg_lg, E);
  }
  k_zero<<<cdiv(N*8,256),256,0,stream>>>(sum_in, N*8);
  k_edge_f<<<cdiv(E,256),256,0,stream>>>(m, w, en, dst, Rs, lf, sum_in, E);
  k_beliefs<<<cdiv(N,256),256,0,stream>>>(log_phi, sum_in, msg_lg, (float*)d_out, N);
}
